// RegressionFNP2_592705487561
// MI455X (gfx1250) — matrix operand order repaired and checked by reading, not compiled, not run on hardware
//
#include <hip/hip_runtime.h>
#include <hip/hip_bf16.h>

typedef __attribute__((ext_vector_type(16))) _Float16 v16h;
typedef __attribute__((ext_vector_type(8)))  _Float16 v8h;
typedef __attribute__((ext_vector_type(8)))  float    v8f;

#define NRR 4096
#define NMM 8192
#define NTOT 12288
#define DXX 64
#define DHH 256

// ---- WMMA 16x16x32 f16 fragment index maps (ISA 7.12.2, wave32) ----
// A (16x32 f16): lane L holds row M=L&15; element h -> K = (h&7) + 16*(h>>3) + 8*(L>>4)
// B (32x16 f16): lane L holds col N=L&15; element h -> K = (h&7) + 16*(h>>3) + 8*(L>>4)
// C/D (16x16 f32): element r -> row M = r + 8*(lane>=16), col N = lane&15.
__device__ __forceinline__ int a_kmap(int h, int lane) {
    return (h & 7) + ((h >> 3) << 4) + ((lane >> 4) << 3);
}

// Vectorized fragment loaders (2x b128 per fragment instead of 16 scalar loads)
__device__ __forceinline__ v16h load_a_f16(const _Float16* row, int k0, int lane) {
    int s0 = k0 + ((lane >> 4) << 3);
    v8h r0 = *(const v8h*)(row + s0);
    v8h r1 = *(const v8h*)(row + s0 + 16);
    v16h a;
#pragma unroll
    for (int h = 0; h < 8; ++h) { a[h] = r0[h]; a[h + 8] = r1[h]; }
    return a;
}
__device__ __forceinline__ v16h load_a_f32(const float* row, int k0, int lane) {
    int s0 = k0 + ((lane >> 4) << 3);
    v8f r0 = *(const v8f*)(row + s0);
    v8f r1 = *(const v8f*)(row + s0 + 16);
    v16h a;
#pragma unroll
    for (int h = 0; h < 8; ++h) { a[h] = (_Float16)r0[h]; a[h + 8] = (_Float16)r1[h]; }
    return a;
}
// B from f16 transposed weights (col-major: col n is contiguous over K)
__device__ __forceinline__ v16h load_b_f16t(const _Float16* col, int k0, int lane) {
    int s0 = k0 + ((lane >> 4) << 3);
    v8h r0 = *(const v8h*)(col + s0);
    v8h r1 = *(const v8h*)(col + s0 + 16);
    v16h b;
#pragma unroll
    for (int h = 0; h < 8; ++h) { b[h] = r0[h]; b[h + 8] = r1[h]; }
    return b;
}

__device__ __forceinline__ float lognorm(float x, float mean, float ls) {
    float d = (x - mean) * expf(-ls);
    return -0.9189385332046727f - ls - 0.5f * d * d;
}

// ---------------- weight transpose to f16 (N x Kpad), zero-pad K ----------------
__global__ __launch_bounds__(256) void k_transpose(const float* src, _Float16* dst,
                                                   int K, int N, int Kpad) {
    int idx = blockIdx.x * blockDim.x + threadIdx.x;
    if (idx >= N * Kpad) return;
    int n = idx / Kpad, k = idx % Kpad;
    dst[(size_t)n * Kpad + k] = (_Float16)(k < K ? src[(size_t)k * N + n] : 0.f);
}

// ---------------- sR = mean(XR, axis=0) ----------------
__global__ __launch_bounds__(256) void k_sR(const float* XR, float* sR) {
    __shared__ float red[256];
    int c = blockIdx.x;
    float s = 0.f;
    for (int r = threadIdx.x; r < NRR; r += 256) s += XR[r * DXX + c];
    red[threadIdx.x] = s; __syncthreads();
    for (int o = 128; o > 0; o >>= 1) {
        if (threadIdx.x < o) red[threadIdx.x] += red[threadIdx.x + o];
        __syncthreads();
    }
    if (threadIdx.x == 0) sR[c] = red[0] / (float)NRR;
}

// ---------------- H = relu(X_all @ W1 + b1), stored f16 ----------------
__global__ __launch_bounds__(256) void k_gemm_H(const float* XR, const float* XM,
                                                const _Float16* W1t, const float* b1,
                                                _Float16* Hh) {
    int wave = (blockIdx.x * blockDim.x + threadIdx.x) >> 5;
    int lane = threadIdx.x & 31;
    int mt = wave >> 4, nt = wave & 15;
    int m0 = mt * 16, n0 = nt * 16;
    int M = lane & 15, N = lane & 15;
    int row = m0 + M;
    const float* xsrc = (row < NRR) ? (XR + (size_t)row * DXX)
                                    : (XM + (size_t)(row - NRR) * DXX);
    const _Float16* wcol = W1t + (size_t)(n0 + N) * DXX;
    v8f acc = {};
    for (int k0 = 0; k0 < DXX; k0 += 32) {
        v16h a = load_a_f32(xsrc, k0, lane);
        v16h b = load_b_f16t(wcol, k0, lane);
        acc = __builtin_amdgcn_wmma_f32_16x16x32_f16(false, a, false, b, (short)0, acc, false, false);
    }
    for (int r = 0; r < 8; ++r) {
        int Mr = r + ((lane >> 4) << 3);
        float v = acc[r] + b1[n0 + N];
        v = v > 0.f ? v : 0.f;
        Hh[(size_t)(m0 + Mr) * DHH + n0 + N] = (_Float16)v;
    }
}

// ---------------- P = H @ [Wu | Wz] + [bu | bz]  (12288 x 32 f32) ----------------
__global__ __launch_bounds__(256) void k_gemm_PU(const _Float16* Hh,
                                                 const _Float16* Wut, const float* bu,
                                                 const _Float16* Wzt, const float* bz,
                                                 float* P) {
    int wave = (blockIdx.x * blockDim.x + threadIdx.x) >> 5;
    int lane = threadIdx.x & 31;
    int mt = wave >> 1, nt = wave & 1;
    int m0 = mt * 16;
    const _Float16* Wt = nt ? Wzt : Wut;
    const float* bias = nt ? bz : bu;
    int M = lane & 15, N = lane & 15;
    const _Float16* arow = Hh + (size_t)(m0 + M) * DHH;
    const _Float16* wcol = Wt + (size_t)N * DHH;
    v8f acc = {};
    for (int k0 = 0; k0 < DHH; k0 += 32) {
        v16h a = load_a_f16(arow, k0, lane);
        v16h b = load_b_f16t(wcol, k0, lane);
        acc = __builtin_amdgcn_wmma_f32_16x16x32_f16(false, a, false, b, (short)0, acc, false, false);
    }
    for (int r = 0; r < 8; ++r) {
        int Mr = r + ((lane >> 4) << 3);
        P[(size_t)(m0 + Mr) * 32 + nt * 16 + N] = acc[r] + bias[N];
    }
}

// ---------------- u, z, |u|^2, and f16 final_rep (96-wide, zero-padded) ----------------
__global__ __launch_bounds__(256) void k_uz(const float* P, const float* eps_u,
                                            const float* eps_z, const float* sR,
                                            float* u, float* z, float* unorm2,
                                            _Float16* rep) {
    int i = blockIdx.x * blockDim.x + threadIdx.x;
    if (i >= NTOT) return;
    _Float16* r = rep + (size_t)i * 96;
    for (int c = 0; c < 64; ++c) r[c] = (_Float16)sR[c];
    const v8f pm  = *(const v8f*)(P + (size_t)i * 32);
    const v8f pl  = *(const v8f*)(P + (size_t)i * 32 + 8);
    const v8f qm  = *(const v8f*)(P + (size_t)i * 32 + 16);
    const v8f ql  = *(const v8f*)(P + (size_t)i * 32 + 24);
    const v8f eu  = *(const v8f*)(eps_u + (size_t)i * 8);
    const v8f ez  = *(const v8f*)(eps_z + (size_t)i * 8);
    v8f uv, zv;
    float s2 = 0.f;
#pragma unroll
    for (int d = 0; d < 8; ++d) {
        uv[d] = pm[d] + eu[d] * expf(pl[d]);
        zv[d] = qm[d] + ez[d] * expf(ql[d]);
        s2 += uv[d] * uv[d];
        r[64 + d] = (_Float16)zv[d];
        r[72 + d] = (_Float16)uv[d];
    }
    *(v8f*)(u + (size_t)i * 8) = uv;
    *(v8f*)(z + (size_t)i * 8) = zv;
    for (int c = 80; c < 96; ++c) r[c] = (_Float16)0.f;
    unorm2[i] = s2;
}

// ---------------- order_z(uR) ----------------
__global__ __launch_bounds__(256) void k_order(const float* u, float* order) {
    int i = blockIdx.x * blockDim.x + threadIdx.x;
    if (i >= NRR) return;
    const v8f uv = *(const v8f*)(u + (size_t)i * 8);
    float s = 0.f;
#pragma unroll
    for (int d = 0; d < 8; ++d)
        s += logf(0.5f + 0.5f * erff(uv[d] * 0.7071067811865476f));
    order[i] = s;
}

// ---------------- stable rank (== argsort(argsort)) via O(N^2) count ----------------
__global__ __launch_bounds__(256) void k_rank(const float* order, int* rank) {
    __shared__ int red[256];
    int i = blockIdx.x;
    float oi = order[i];
    int cnt = 0;
    for (int j = threadIdx.x; j < NRR; j += 256) {
        float oj = order[j];
        cnt += (oj < oi) || (oj == oi && j < i);
    }
    red[threadIdx.x] = cnt; __syncthreads();
    for (int o = 128; o > 0; o >>= 1) {
        if (threadIdx.x < o) red[threadIdx.x] += red[threadIdx.x + o];
        __syncthreads();
    }
    if (threadIdx.x == 0) rank[i] = red[0];
}

// ---- Ct (16 x 4096 f16, transposed): row c is contiguous over j ----
__global__ __launch_bounds__(256) void k_cmat(const float* P, const float* yR,
                                              const float* Wy, const float* by,
                                              _Float16* Ct) {
    int j = blockIdx.x * blockDim.x + threadIdx.x;
    if (j >= NRR) return;
    float y = yR[j];
    for (int c = 0; c < 8; ++c) {
        Ct[(size_t)c * NRR + j]       = (_Float16)(y * Wy[c]     + by[c]     + P[(size_t)j * 32 + 16 + c]);
        Ct[(size_t)(c + 8) * NRR + j] = (_Float16)(y * Wy[8 + c] + by[8 + c] + P[(size_t)j * 32 + 24 + c]);
    }
}

// ---------------- fused: G/A logits -> relaxed bernoulli -> rowsum -> WMMA @ C -> pz ----------------
__global__ __launch_bounds__(256) void k_ga(const float* u, const float* unorm2,
                                            const int* rank,
                                            const float* unif_G, const float* unif_A,
                                            const _Float16* Ct,
                                            const float* g_logscale, float* pz) {
    __shared__ float s_ui[16][8];
    __shared__ float s_ui2[16];
    __shared__ int   s_ri[16];
    __shared__ float s_ur[8][32][8];
    __shared__ float s_ur2[8][32];
    __shared__ int   s_rj[8][32];
    __shared__ float s_acc[8][32][8];
    __shared__ float s_rs[8][16];

    int tid = threadIdx.x, lane = tid & 31, wave = tid >> 5;
    int i0 = blockIdx.x * 16;
    if (tid < 128) { int m = tid >> 3, d = tid & 7; s_ui[m][d] = u[(size_t)(i0 + m) * 8 + d]; }
    if (tid < 16) {
        s_ui2[tid] = unorm2[i0 + tid];
        s_ri[tid]  = (i0 + tid) < NRR ? rank[i0 + tid] : 0;
    }
    __syncthreads();

    const float inv_scale = expf(-g_logscale[0]);
    const bool  isG = (i0 < NRR);          // NR % 16 == 0: tiles never straddle
    int   M = lane & 15;
    float uiM[8];
    for (int d = 0; d < 8; ++d) uiM[d] = s_ui[M][d];
    float ui2M = s_ui2[M];
    int   riM = s_ri[M];
    const float* uArow = unif_A + (size_t)(i0 + M - NRR) * NRR;   // valid only when !isG
    const float* uGrow = unif_G + (size_t)riM * NRR;
    const _Float16* ccol = Ct + (size_t)(lane & 15) * NRR;

    v8f   acc = {};
    float rsum = 0.f;

    for (int t = 0; t < 16; ++t) {
        int j0 = (wave + 8 * t) * 32;
        {   // per-wave stage of this chunk's 32 uR rows
            int j = j0 + lane;
            v8f ur = *(const v8f*)(u + (size_t)j * 8);
            for (int d = 0; d < 8; ++d) s_ur[wave][lane][d] = ur[d];
            s_ur2[wave][lane] = unorm2[j];
            s_rj[wave][lane]  = rank[j];
        }
        __syncthreads();

        // uniform randoms: A-part is contiguous (2x 32B loads); G-part is a rank gather
        float uval[16];
        int   rjv[16];
        if (isG) {
#pragma unroll
            for (int h = 0; h < 16; ++h) {
                int K = a_kmap(h, lane);
                rjv[h]  = s_rj[wave][K];
                uval[h] = uGrow[rjv[h]];
            }
        } else {
            int s0 = j0 + ((lane >> 4) << 3);
            v8f q0 = *(const v8f*)(uArow + s0);
            v8f q1 = *(const v8f*)(uArow + s0 + 16);
#pragma unroll
            for (int h = 0; h < 8; ++h) { uval[h] = q0[h]; uval[h + 8] = q1[h]; }
        }

        v16h a;
#pragma unroll
        for (int h = 0; h < 16; ++h) {
            int K = a_kmap(h, lane);
            float dot = 0.f;
            for (int d = 0; d < 8; ++d) dot += uiM[d] * s_ur[wave][K][d];
            float d2 = ui2M + s_ur2[wave][K] - 2.f * dot;
            d2 = d2 > 0.f ? d2 : 0.f;
            float logp = -0.5f * d2 * inv_scale;
            float L;
            if (logp > -0.6931471805599453f)
                L = -logf(fmaxf(expm1f(-logp), 1e-20f));
            else
                L = logp - log1pf(-expf(logp));
            float uu = uval[h];
            float mask = isG ? (rjv[h] > riM ? 1.f : 0.f) : 1.f;
            float tt = (L + logf(uu) - log1pf(-uu)) * (1.f / 0.3f);
            float g = 1.f / (1.f + expf(-tt));
            g *= mask;
            rsum += g;
            a[h] = (_Float16)g;
        }
        v16h b = load_b_f16t(ccol, j0, lane);
        acc = __builtin_amdgcn_wmma_f32_16x16x32_f16(false, a, false, b, (short)0, acc, false, false);
        __syncthreads();
    }

    // rows split across lanes L and L^16: combine rowsum within wave
    rsum += __shfl_xor(rsum, 16, 32);
    if (lane < 16) s_rs[wave][lane] = rsum;
    for (int r = 0; r < 8; ++r) s_acc[wave][lane][r] = acc[r];
    __syncthreads();

    if (wave == 0) {
        int N = lane & 15;
        for (int r = 0; r < 8; ++r) {
            int Mr = r + ((lane >> 4) << 3);
            float tot = 0.f, rs = 0.f;
            for (int w = 0; w < 8; ++w) { tot += s_acc[w][lane][r]; rs += s_rs[w][Mr]; }
            pz[(size_t)(i0 + Mr) * 16 + N] = tot / (rs + 1e-8f);
        }
    }
}

// ---------------- O = relu(rep @ Wo1 + bo1), stored f16 ----------------
__global__ __launch_bounds__(256) void k_gemm_O(const _Float16* rep, const _Float16* Wo1t,
                                                const float* bo1, _Float16* Oh) {
    int wave = (blockIdx.x * blockDim.x + threadIdx.x) >> 5;
    int lane = threadIdx.x & 31;
    int mt = wave >> 4, nt = wave & 15;
    int m0 = mt * 16, n0 = nt * 16;
    int M = lane & 15, N = lane & 15;
    const _Float16* arow = rep + (size_t)(m0 + M) * 96;
    const _Float16* wcol = Wo1t + (size_t)(n0 + N) * 96;
    v8f acc = {};
    for (int k0 = 0; k0 < 96; k0 += 32) {
        v16h a = load_a_f16(arow, k0, lane);
        v16h b = load_b_f16t(wcol, k0, lane);
        acc = __builtin_amdgcn_wmma_f32_16x16x32_f16(false, a, false, b, (short)0, acc, false, false);
    }
    for (int r = 0; r < 8; ++r) {
        int Mr = r + ((lane >> 4) << 3);
        float v = acc[r] + bo1[n0 + N];
        v = v > 0.f ? v : 0.f;
        Oh[(size_t)(m0 + Mr) * DHH + n0 + N] = (_Float16)v;
    }
}

// ---------------- head: mean_y, logstd_y, log_py sums ----------------
__global__ __launch_bounds__(256) void k_head(const _Float16* Oh, const float* Wo2,
                                              const float* bo2, const float* yR,
                                              const float* yM, float* out, float* acc) {
    int i = blockIdx.x * blockDim.x + threadIdx.x;
    if (i >= NTOT) return;
    float m = bo2[0], l = bo2[1];
    const v8h* op = (const v8h*)(Oh + (size_t)i * DHH);
    for (int k8 = 0; k8 < DHH / 8; ++k8) {
        v8h hv = op[k8];
#pragma unroll
        for (int j = 0; j < 8; ++j) {
            float h = (float)hv[j];
            m += h * Wo2[(k8 * 8 + j) * 2 + 0];
            l += h * Wo2[(k8 * 8 + j) * 2 + 1];
        }
    }
    float sp = (l > 20.f) ? l : log1pf(expf(l));
    float logstd = logf(0.1f + 0.9f * sp);
    out[1 + i] = m;
    out[1 + NTOT + i] = logstd;
    float y = (i < NRR) ? yR[i] : yM[i - NRR];
    atomicAdd(&acc[i < NRR ? 0 : 1], lognorm(y, m, logstd));
}

// ---------------- pqz sums ----------------
__global__ __launch_bounds__(256) void k_pqz(const float* z, const float* pz,
                                             const float* P, float* acc) {
    int i = blockIdx.x * blockDim.x + threadIdx.x;
    if (i >= NTOT) return;
    const v8f zv = *(const v8f*)(z + (size_t)i * 8);
    const v8f pzm = *(const v8f*)(pz + (size_t)i * 16);
    const v8f pzl = *(const v8f*)(pz + (size_t)i * 16 + 8);
    const v8f qm = *(const v8f*)(P + (size_t)i * 32 + 16);
    const v8f ql = *(const v8f*)(P + (size_t)i * 32 + 24);
    float s = 0.f;
#pragma unroll
    for (int c = 0; c < 8; ++c)
        s += lognorm(zv[c], pzm[c], pzl[c]) - lognorm(zv[c], qm[c], ql[c]);
    atomicAdd(&acc[i < NRR ? 2 : 3], s);
}

__global__ void k_init(float* acc) { if (threadIdx.x < 4) acc[threadIdx.x] = 0.f; }

__global__ void k_loss(const float* acc, float* out) {
    if (threadIdx.x == 0)
        out[0] = -((acc[0] + acc[2]) + (acc[1] + acc[3])) / (float)NMM;
}

extern "C" void kernel_launch(void* const* d_in, const int* in_sizes, int n_in,
                              void* d_out, int out_size, void* d_ws, size_t ws_size,
                              hipStream_t stream) {
    const float* XR   = (const float*)d_in[0];
    const float* yR   = (const float*)d_in[1];
    const float* XM   = (const float*)d_in[2];
    const float* yM   = (const float*)d_in[3];
    const float* gls  = (const float*)d_in[4];
    const float* W1   = (const float*)d_in[5];
    const float* b1   = (const float*)d_in[6];
    const float* Wu   = (const float*)d_in[7];
    const float* bu   = (const float*)d_in[8];
    const float* Wz   = (const float*)d_in[9];
    const float* bz   = (const float*)d_in[10];
    const float* Wy   = (const float*)d_in[11];
    const float* by   = (const float*)d_in[12];
    const float* Wo1  = (const float*)d_in[13];
    const float* bo1  = (const float*)d_in[14];
    const float* Wo2  = (const float*)d_in[15];
    const float* bo2  = (const float*)d_in[16];
    const float* eps_u = (const float*)d_in[17];
    const float* eps_z = (const float*)d_in[18];
    const float* unif_G = (const float*)d_in[19];
    const float* unif_A = (const float*)d_in[20];
    float* out = (float*)d_out;

    char* ws = (char*)d_ws;
    size_t off = 0;
    auto alloc = [&](size_t bytes) {
        void* p = ws + off;
        off = (off + bytes + 255) & ~(size_t)255;
        return p;
    };
    _Float16* Hh    = (_Float16*)alloc((size_t)NTOT * DHH * 2);
    float*    P     = (float*)   alloc((size_t)NTOT * 32 * 4);
    float*    u     = (float*)   alloc((size_t)NTOT * 8 * 4);
    float*    z     = (float*)   alloc((size_t)NTOT * 8 * 4);
    float*    un2   = (float*)   alloc((size_t)NTOT * 4);
    float*    sR    = (float*)   alloc(64 * 4);
    float*    order = (float*)   alloc((size_t)NRR * 4);
    int*      rank  = (int*)     alloc((size_t)NRR * 4);
    _Float16* Ct    = (_Float16*)alloc((size_t)16 * NRR * 2);
    float*    pz    = (float*)   alloc((size_t)NTOT * 16 * 4);
    _Float16* rep   = (_Float16*)alloc((size_t)NTOT * 96 * 2);
    _Float16* Oh    = (_Float16*)alloc((size_t)NTOT * DHH * 2);
    _Float16* W1t   = (_Float16*)alloc((size_t)DHH * DXX * 2);   // 256 x 64
    _Float16* Wut   = (_Float16*)alloc((size_t)16 * DHH * 2);    // 16 x 256
    _Float16* Wzt   = (_Float16*)alloc((size_t)16 * DHH * 2);    // 16 x 256
    _Float16* Wo1t  = (_Float16*)alloc((size_t)DHH * 96 * 2);    // 256 x 96 (K 80->96 zero-pad)
    float*    acc   = (float*)   alloc(8 * 4);

    k_init<<<1, 32, 0, stream>>>(acc);
    k_transpose<<<(DHH * DXX + 255) / 256, 256, 0, stream>>>(W1, W1t, DXX, DHH, DXX);
    k_transpose<<<(16 * DHH + 255) / 256, 256, 0, stream>>>(Wu, Wut, DHH, 16, DHH);
    k_transpose<<<(16 * DHH + 255) / 256, 256, 0, stream>>>(Wz, Wzt, DHH, 16, DHH);
    k_transpose<<<(DHH * 96 + 255) / 256, 256, 0, stream>>>(Wo1, Wo1t, 80, DHH, 96);
    k_sR<<<64, 256, 0, stream>>>(XR, sR);
    k_gemm_H<<<(NTOT / 16) * 16 / 8, 256, 0, stream>>>(XR, XM, W1t, b1, Hh);      // 1536 blocks
    k_gemm_PU<<<(NTOT / 16) * 2 / 8, 256, 0, stream>>>(Hh, Wut, bu, Wzt, bz, P);  // 192 blocks
    k_uz<<<(NTOT + 255) / 256, 256, 0, stream>>>(P, eps_u, eps_z, sR, u, z, un2, rep);
    k_order<<<(NRR + 255) / 256, 256, 0, stream>>>(u, order);
    k_rank<<<NRR, 256, 0, stream>>>(order, rank);
    k_cmat<<<(NRR + 255) / 256, 256, 0, stream>>>(P, yR, Wy, by, Ct);
    k_ga<<<NTOT / 16, 256, 0, stream>>>(u, un2, rank, unif_G, unif_A, Ct, gls, pz);
    k_gemm_O<<<(NTOT / 16) * 16 / 8, 256, 0, stream>>>(rep, Wo1t, bo1, Oh);
    k_head<<<(NTOT + 255) / 256, 256, 0, stream>>>(Oh, Wo2, bo2, yR, yM, out, acc);
    k_pqz<<<(NTOT + 255) / 256, 256, 0, stream>>>(z, pz, P, acc);
    k_loss<<<1, 32, 0, stream>>>(acc, out);
}
